// ProposalModuleRefine_23063974380063
// MI455X (gfx1250) — compile-verified, matrix operand order checked by reading, not run on hardware
//
#include <hip/hip_runtime.h>

#define B_    8
#define NP_   256
#define NPRIM 1024
#define NS_   18

typedef __attribute__((ext_vector_type(16))) _Float16 v16h;
typedef __attribute__((ext_vector_type(8)))  float    v8f;
typedef __attribute__((ext_vector_type(4)))  unsigned u32x4;

union V16H  { v16h v; u32x4 q[2]; };
union HPAIR { _Float16 h[2]; unsigned u; };

__constant__ float SURF_COEF_c[6][3] = {
  {0,0,.5f},{0,0,-.5f},{0,.5f,0},{0,-.5f,0},{.5f,0,0},{-.5f,0,0}};
__constant__ float LINE_COEF_c[12][3] = {
  {.5f,0,.5f},{-.5f,0,.5f},{0,.5f,.5f},{0,-.5f,.5f},
  {.5f,0,-.5f},{-.5f,0,-.5f},{0,.5f,-.5f},{0,-.5f,-.5f},
  {.5f,.5f,0},{.5f,-.5f,0},{-.5f,.5f,0},{-.5f,-.5f,0}};

// ---------------------------------------------------------------- prep ----

// obj_size[b,n,3] = mean_size[argmax(sem)] + size_res[b,n,argmax,:]
__global__ void k_objsize(const float* __restrict__ sem,
                          const float* __restrict__ sres,
                          const float* __restrict__ msz,
                          float* __restrict__ osz) {
  int t = blockIdx.x * blockDim.x + threadIdx.x;
  if (t >= B_ * NP_) return;
  const float* s = sem + (size_t)t * NS_;
  int cls = 0; float best = s[0];
  for (int i = 1; i < NS_; ++i) { float v = s[i]; if (v > best) { best = v; cls = i; } }
  for (int k = 0; k < 3; ++k)
    osz[t * 3 + k] = msz[cls * 3 + k] + sres[((size_t)t * NS_ + cls) * 3 + k];
}

// surf_pts: [0,1024) push_far(center_z), [1024,2048) push_far(center_xy),
//           [2048,3584) box surface centers
__global__ void k_build_surf(const float* __restrict__ cz, const float* __restrict__ flz,
                             const float* __restrict__ cxy, const float* __restrict__ flxy,
                             const float* __restrict__ ctr, const float* __restrict__ osz,
                             float* __restrict__ pts) {
  int t = blockIdx.x * blockDim.x + threadIdx.x;
  if (t >= B_ * 3584) return;
  int b = t / 3584, j = t % 3584;
  float p[3];
  if (j < 2048) {
    const float* c; const float* f; int jj;
    if (j < 1024) { c = cz; f = flz; jj = j; } else { c = cxy; f = flxy; jj = j - 1024; }
    float f0 = f[((size_t)b * 2 + 0) * NPRIM + jj];
    float f1 = f[((size_t)b * 2 + 1) * NPRIM + jj];
    float push = (f1 <= f0) ? 100.f : 0.f;           // softmax p1<=0.5 <=> f1<=f0
    for (int k = 0; k < 3; ++k) p[k] = c[((size_t)b * NPRIM + jj) * 3 + k] + push;
  } else {
    int k6 = (j - 2048) >> 8, n = (j - 2048) & 255;
    for (int k = 0; k < 3; ++k)
      p[k] = ctr[((size_t)b * NP_ + n) * 3 + k] +
             SURF_COEF_c[k6][k] * osz[((size_t)b * NP_ + n) * 3 + k];
  }
  for (int k = 0; k < 3; ++k) pts[(size_t)t * 3 + k] = p[k];
}

__global__ void k_build_line(const float* __restrict__ cl, const float* __restrict__ fll,
                             const float* __restrict__ ctr, const float* __restrict__ osz,
                             float* __restrict__ pts) {
  int t = blockIdx.x * blockDim.x + threadIdx.x;
  if (t >= B_ * 4096) return;
  int b = t / 4096, j = t % 4096;
  float p[3];
  if (j < 1024) {
    float f0 = fll[((size_t)b * 2 + 0) * NPRIM + j];
    float f1 = fll[((size_t)b * 2 + 1) * NPRIM + j];
    float push = (f1 <= f0) ? 100.f : 0.f;
    for (int k = 0; k < 3; ++k) p[k] = cl[((size_t)b * NPRIM + j) * 3 + k] + push;
  } else {
    int k12 = (j - 1024) >> 8, n = (j - 1024) & 255;
    for (int k = 0; k < 3; ++k)
      p[k] = ctr[((size_t)b * NP_ + n) * 3 + k] +
             LINE_COEF_c[k12][k] * osz[((size_t)b * NP_ + n) * 3 + k];
  }
  for (int k = 0; k < 3; ++k) pts[(size_t)t * 3 + k] = p[k];
}

// Point-major f16 feature table, Kpad=160. ch0-2 reserved for xyz (zero here).
// surface: ch3..8 indicator(6), ch9..136 features; pad to 160 with zeros.
__global__ void k_feat_surf(const float* __restrict__ fz, const float* __restrict__ fxy,
                            const float* __restrict__ agg, _Float16* __restrict__ dst) {
  int t = blockIdx.x * blockDim.x + threadIdx.x;
  if (t >= B_ * 3584 * 160) return;
  int c = t % 160, j = (t / 160) % 3584, b = t / (160 * 3584);
  float v = 0.f;
  if (c >= 3 && c < 137) {
    int cf = c - 3;                       // 0..133
    if (j < 2048) {
      if (cf >= 6)
        v = (j < 1024) ? fz [((size_t)b * 128 + cf - 6) * NPRIM + j]
                       : fxy[((size_t)b * 128 + cf - 6) * NPRIM + (j - 1024)];
    } else {
      int k6 = (j - 2048) >> 8, n = (j - 2048) & 255;
      v = (cf < 6) ? (cf == k6 ? 1.f : 0.f)
                   : agg[((size_t)b * 128 + cf - 6) * NP_ + n];
    }
  }
  dst[t] = (_Float16)v;
}

__global__ void k_feat_line(const float* __restrict__ fl,
                            const float* __restrict__ agg, _Float16* __restrict__ dst) {
  int t = blockIdx.x * blockDim.x + threadIdx.x;
  if (t >= B_ * 4096 * 160) return;
  int c = t % 160, j = (t / 160) % 4096, b = t / (160 * 4096);
  float v = 0.f;
  if (c >= 3 && c < 146) {
    int cf = c - 3;                       // 0..142
    if (j < 1024) {
      if (cf >= 12) v = fl[((size_t)b * 128 + cf - 12) * NPRIM + j];
    } else {
      int k12 = (j - 1024) >> 8, n = (j - 1024) & 255;
      v = (cf < 12) ? (cf == k12 ? 1.f : 0.f)
                    : agg[((size_t)b * 128 + cf - 12) * NP_ + n];
    }
  }
  dst[t] = (_Float16)v;
}

// Pack W (Kreal x N, row major f32) into WMMA-B per-lane layout, f16.
// dword index ((c*NT+nt)*32+lane)*8+d ; lane n = nt*16+(lane&15),
// k = c*32 + (lane<16?0:8) + (d<4 ? 2d : 16+2(d-4)) (+1 in high half)
__global__ void k_pack(const float* __restrict__ w, int Kreal, int N, int NT,
                       unsigned* __restrict__ dst) {
  int t = blockIdx.x * 256 + threadIdx.x;
  int d = t & 7, lane = (t >> 3) & 31, nt = (t >> 8) % NT, c = t / (NT << 8);
  int n  = nt * 16 + (lane & 15);
  int kb = c * 32 + ((lane < 16) ? 0 : 8) + (d < 4 ? 2 * d : 16 + 2 * (d - 4));
  HPAIR p;
  p.h[0] = (_Float16)((kb     < Kreal && n < N) ? w[(size_t)kb * N + n]       : 0.f);
  p.h[1] = (_Float16)((kb + 1 < Kreal && n < N) ? w[(size_t)(kb + 1) * N + n] : 0.f);
  dst[t] = p.u;
}

// ---------------------------------------------------------- ball query ----
// one wave32 per query: first 32 in-radius indices in point order,
// pad with first hit (or 0 when none)
__global__ __launch_bounds__(128)
void k_group(const float* __restrict__ pts, int P, int qoff, int Q,
             int* __restrict__ idxo) {
  int lane = threadIdx.x & 31, wave = threadIdx.x >> 5;
  int qg = blockIdx.x * 4 + wave;
  int b = qg / Q, qi = qg - b * Q;
  const float* base = pts + (size_t)b * P * 3;
  const float* qp = base + (size_t)(qoff + qi) * 3;
  float qx = qp[0], qy = qp[1], qz = qp[2];
  int* out = idxo + ((size_t)qg << 5);
  int cnt = 0, first = -1;
  for (int s = 0; s < P && cnt < 32; s += 32) {
    int j = s + lane;
    const float* pp = base + (size_t)j * 3;
    float dx = pp[0] - qx, dy = pp[1] - qy, dz = pp[2] - qz;
    bool inr = (dx * dx + dy * dy + dz * dz) < 0.25f;
    unsigned long long msk = __ballot(inr);
    int rank = __popcll(msk & ((1ull << lane) - 1ull));
    if (inr && cnt + rank < 32) out[cnt + rank] = j;
    if (first < 0 && msk) first = s + (__ffsll((long long)msk) - 1);
    cnt += __popcll(msk);
  }
  int fill = (first < 0) ? 0 : first;
  for (int s2 = (cnt < 32 ? cnt : 32) + lane; s2 < 32; s2 += 32) out[s2] = fill;
}

// ------------------------------------------------------------ wmma core ----
// A layout (16x32 f16): lane<16 -> m=lane, k: {kh..kh+7, kh+16..kh+23}, kh=0
//                       lane>=16 -> m=lane-16, kh=8
__device__ __forceinline__ v8f gemm_rowtile(const _Float16* __restrict__ xrow,
                                            const unsigned* __restrict__ Wp,
                                            int KC, int NT, int nt, int lane) {
  v8f acc = {};
  int khalf = (lane < 16) ? 0 : 8;
  for (int c = 0; c < KC; ++c) {
    V16H a, bb;
    const _Float16* ap = xrow + c * 32 + khalf;
    a.q[0] = *(const u32x4*)ap;
    a.q[1] = *(const u32x4*)(ap + 16);
    const u32x4* bp = (const u32x4*)Wp + ((size_t)(c * NT + nt) * 32 + lane) * 2;
    bb.q[0] = bp[0];
    bb.q[1] = bp[1];
    acc = __builtin_amdgcn_wmma_f32_16x16x32_f16(false, a.v, false, bb.v,
                                                 (short)0, acc, false, false);
  }
  return acc;
}

// ------------------------------------------------- fused SA (ball MLP) ----
// 128 thr (4 waves), 2 queries (64 rows) / WG. gather->L1->L2->L3->maxpool.
__global__ __launch_bounds__(128)
void k_sa(const float* __restrict__ pts, int P, int qoff, int Q,
          const _Float16* __restrict__ feat, const int* __restrict__ idx,
          const unsigned* __restrict__ W1, const float* __restrict__ b1,
          const unsigned* __restrict__ W2, const float* __restrict__ b2,
          const unsigned* __restrict__ W3, const float* __restrict__ b3,
          _Float16* __restrict__ comb, int qbase) {
  __shared__ __align__(16) _Float16 Xs[64 * 160];   // 20KB: input, later H2
  __shared__ __align__(16) _Float16 Hs[64 * 128];   // 16KB: H1,   later H3(f32)
  int tid = threadIdx.x, lane = tid & 31, wave = tid >> 5;
  int t0 = blockIdx.x * 2;
  int b = t0 / Q, q0 = t0 % Q;
  { // gather: 2 threads per row, 80 f16 each
    int r = tid >> 1, half = tid & 1;
    int ql = r >> 5, s = r & 31;
    int q = q0 + ql;
    int j = idx[((size_t)(b * Q + q) << 5) + s];
    const u32x4* src = (const u32x4*)(feat + ((size_t)b * P + j) * 160) + half * 10;
    u32x4* dst = (u32x4*)(Xs + r * 160) + half * 10;
    for (int i = 0; i < 10; ++i) dst[i] = src[i];
    if (half == 0) {   // g_xyz = (p - q)/radius, radius=0.5
      const float* qp = pts + ((size_t)b * P + qoff + q) * 3;
      const float* pp = pts + ((size_t)b * P + j) * 3;
      Xs[r * 160 + 0] = (_Float16)((pp[0] - qp[0]) * 2.f);
      Xs[r * 160 + 1] = (_Float16)((pp[1] - qp[1]) * 2.f);
      Xs[r * 160 + 2] = (_Float16)((pp[2] - qp[2]) * 2.f);
    }
  }
  __syncthreads();
  int m = lane & 15, khalf = (lane < 16) ? 0 : 8;
  // L1: 160 -> 128
  for (int nt = 0; nt < 8; ++nt) {
    v8f acc = gemm_rowtile(Xs + (wave * 16 + m) * 160, W1, 5, 8, nt, lane);
    int n = nt * 16 + m; float bv = b1[n];
    for (int i = 0; i < 8; ++i)
      Hs[(wave * 16 + khalf + i) * 128 + n] = (_Float16)fmaxf(acc[i] + bv, 0.f);
  }
  __syncthreads();
  // L2: 128 -> 64 (into Xs region)
  _Float16* X2 = Xs;
  for (int nt = 0; nt < 4; ++nt) {
    v8f acc = gemm_rowtile(Hs + (wave * 16 + m) * 128, W2, 4, 4, nt, lane);
    int n = nt * 16 + m; float bv = b2[n];
    for (int i = 0; i < 8; ++i)
      X2[(wave * 16 + khalf + i) * 64 + n] = (_Float16)fmaxf(acc[i] + bv, 0.f);
  }
  __syncthreads();
  // L3: 64 -> 32 (f32 into Hs region)
  float* H3 = (float*)Hs;
  for (int nt = 0; nt < 2; ++nt) {
    v8f acc = gemm_rowtile(X2 + (wave * 16 + m) * 64, W3, 2, 2, nt, lane);
    int n = nt * 16 + m; float bv = b3[n];
    for (int i = 0; i < 8; ++i)
      H3[(wave * 16 + khalf + i) * 32 + n] = fmaxf(acc[i] + bv, 0.f);
  }
  __syncthreads();
  if (tid < 64) { // maxpool over 32 samples
    int ql = tid >> 5, ch = tid & 31;
    float mx = H3[(ql * 32) * 32 + ch];
    for (int s = 1; s < 32; ++s) mx = fmaxf(mx, H3[(ql * 32 + s) * 32 + ch]);
    comb[((size_t)b * 4608 + qbase + q0 + ql) * 32 + ch] = (_Float16)mx;
  }
}

// ----------------------------------------------------- generic WMMA GEMM ----
// 128 thr (4 waves), 64 rows/WG. Input rows remapped g -> (g/rpb)*bstr+base+g%rpb.
// epi: 0 f16 rows +ReLU | 2 f32 out rowmajor | 3 f32 out chmajor(256 cols)
//      4 sf/lf reshape scatter +ReLU (p0 base, p1 mul, p2 rows_per_b)
//      5 ReLU + agg_feat residual -> f16 rows(128)
__global__ __launch_bounds__(128)
void k_gemm(const _Float16* __restrict__ X, int Kpad, int KC,
            const unsigned* __restrict__ Wp, int NT,
            const float* __restrict__ bias, int Nreal,
            int rpb_in, int bstr_in, int base_in,
            int epi, float* __restrict__ outf, _Float16* __restrict__ outh,
            const float* __restrict__ aux, int p0, int p1, int p2) {
  int lane = threadIdx.x & 31, wave = threadIdx.x >> 5;
  int g0 = blockIdx.x * 64 + wave * 16;
  int ga = g0 + (lane & 15);
  int rb = ga / rpb_in, rq = ga - rb * rpb_in;
  const _Float16* xrow = X + (size_t)(rb * bstr_in + base_in + rq) * Kpad;
  int wout = NT * 16;
  int khalf = (lane < 16) ? 0 : 8;
  for (int nt = 0; nt < NT; ++nt) {
    v8f acc = gemm_rowtile(xrow, Wp, KC, NT, nt, lane);
    int n = nt * 16 + (lane & 15);
    float bv = (n < Nreal) ? bias[n] : 0.f;
    for (int i = 0; i < 8; ++i) {
      int gg = g0 + khalf + i;
      float v = acc[i] + bv;
      if (epi == 0) {
        outh[(size_t)gg * wout + n] = (_Float16)fmaxf(v, 0.f);
      } else if (epi == 2) {
        if (n < Nreal) outf[(size_t)gg * Nreal + n] = v;
      } else if (epi == 3) {
        int bb = gg >> 8, nn = gg & 255;
        if (n < Nreal) outf[((size_t)bb * Nreal + n) * 256 + nn] = v;
      } else if (epi == 4) {
        int bb = gg / p2, qq = gg - bb * p2;
        int k = qq >> 8, nn = qq & 255;
        outh[((size_t)bb * 256 + nn) * 576 + p0 + n * p1 + k] =
            (_Float16)fmaxf(v, 0.f);
      } else if (epi == 5) {
        int bb = gg >> 8, nn = gg & 255;
        float r = fmaxf(v, 0.f) + aux[((size_t)bb * 128 + n) * 256 + nn];
        outh[(size_t)gg * 128 + n] = (_Float16)r;
      }
    }
  }
}

// ------------------------------------------------------------------ host ----
extern "C" void kernel_launch(void* const* d_in, const int* in_sizes, int n_in,
                              void* d_out, int out_size, void* d_ws, size_t ws_size,
                              hipStream_t stream) {
  (void)in_sizes; (void)n_in; (void)out_size; (void)ws_size;
  const float* center_z  = (const float*)d_in[0];
  const float* feat_z    = (const float*)d_in[1];
  const float* center_xy = (const float*)d_in[2];
  const float* feat_xy   = (const float*)d_in[3];
  const float* center_l  = (const float*)d_in[4];
  const float* feat_l    = (const float*)d_in[5];
  const float* flag_z    = (const float*)d_in[6];
  const float* flag_xy   = (const float*)d_in[7];
  const float* flag_l    = (const float*)d_in[8];
  const float* center    = (const float*)d_in[9];
  const float* size_res  = (const float*)d_in[10];
  const float* sem       = (const float*)d_in[11];
  const float* agg       = (const float*)d_in[12];
  const float* msz       = (const float*)d_in[13];
  auto W = [&](int i) { return (const float*)d_in[i]; };
  float* out = (float*)d_out;

  char* w = (char*)d_ws; size_t off = 0;
  auto alloc = [&](size_t bytes) -> void* {
    off = (off + 255) & ~(size_t)255;
    void* p = w + off; off += bytes; return p;
  };
  float*    osz  = (float*)alloc(2048ul * 3 * 4);
  float*    spts = (float*)alloc(8ul * 3584 * 3 * 4);
  float*    lpts = (float*)alloc(8ul * 4096 * 3 * 4);
  _Float16* fs   = (_Float16*)alloc(8ul * 3584 * 160 * 2);
  _Float16* flm  = (_Float16*)alloc(8ul * 4096 * 160 * 2);
  int*      idxs = (int*)alloc(8ul * 1536 * 32 * 4);
  int*      idxl = (int*)alloc(8ul * 3072 * 32 * 4);
  _Float16* comb = (_Float16*)alloc(8ul * 4608 * 32 * 2);
  unsigned* w1s  = (unsigned*)alloc(5ul * 8 * 256 * 4);
  unsigned* w2s  = (unsigned*)alloc(4ul * 4 * 256 * 4);
  unsigned* w3s  = (unsigned*)alloc(2ul * 2 * 256 * 4);
  unsigned* w1l  = (unsigned*)alloc(5ul * 8 * 256 * 4);
  unsigned* w2l  = (unsigned*)alloc(4ul * 4 * 256 * 4);
  unsigned* w3l  = (unsigned*)alloc(2ul * 2 * 256 * 4);
  unsigned* wm1  = (unsigned*)alloc(2ul * 256 * 4);
  unsigned* wm2  = (unsigned*)alloc(1ul * 256 * 4);
  unsigned* wms1 = (unsigned*)alloc(2ul * 256 * 4);
  unsigned* wms2 = (unsigned*)alloc(1ul * 256 * 4);
  unsigned* wsa1 = (unsigned*)alloc(2ul * 256 * 4);
  unsigned* wsa2 = (unsigned*)alloc(2ul * 256 * 4);
  unsigned* wla1 = (unsigned*)alloc(2ul * 256 * 4);
  unsigned* wla2 = (unsigned*)alloc(2ul * 256 * 4);
  unsigned* wr1  = (unsigned*)alloc(18ul * 8 * 256 * 4);
  unsigned* wr2  = (unsigned*)alloc(4ul * 8 * 256 * 4);
  unsigned* wr3  = (unsigned*)alloc(4ul * 8 * 256 * 4);
  unsigned* wr4  = (unsigned*)alloc(4ul * 8 * 256 * 4);
  _Float16* hbuf = (_Float16*)alloc(36864ul * 32 * 2);
  _Float16* xfin = (_Float16*)alloc(2048ul * 576 * 2);
  _Float16* xa   = (_Float16*)alloc(2048ul * 128 * 2);
  _Float16* xb   = (_Float16*)alloc(2048ul * 128 * 2);

  // prep
  k_objsize<<<8, 256, 0, stream>>>(sem, size_res, msz, osz);
  k_build_surf<<<112, 256, 0, stream>>>(center_z, flag_z, center_xy, flag_xy,
                                        center, osz, spts);
  k_build_line<<<128, 256, 0, stream>>>(center_l, flag_l, center, osz, lpts);
  k_feat_surf<<<17920, 256, 0, stream>>>(feat_z, feat_xy, agg, fs);
  k_feat_line<<<20480, 256, 0, stream>>>(feat_l, agg, flm);

  // pack all weights into WMMA-B layout
  auto pack = [&](const float* src, int Kreal, int N, int KC, int NT, unsigned* dst) {
    k_pack<<<KC * NT, 256, 0, stream>>>(src, Kreal, N, NT, dst);
  };
  pack(W(14), 137, 128, 5, 8, w1s);  pack(W(16), 128, 64, 4, 4, w2s);
  pack(W(18),  64,  32, 2, 2, w3s);
  pack(W(20), 143, 128, 5, 8, w1l);  pack(W(22), 128, 64, 4, 4, w2l);
  pack(W(24),  64,  32, 2, 2, w3l);
  pack(W(26), 32, 32, 1, 2, wm1);    pack(W(28), 32,  2, 1, 1, wm2);
  pack(W(30), 32, 32, 1, 2, wms1);   pack(W(32), 32,  2, 1, 1, wms2);
  pack(W(34), 32, 32, 1, 2, wsa1);   pack(W(36), 32, 32, 1, 2, wsa2);
  pack(W(38), 32, 32, 1, 2, wla1);   pack(W(40), 32, 32, 1, 2, wla2);
  pack(W(42), 576, 128, 18, 8, wr1); pack(W(44), 128, 128, 4, 8, wr2);
  pack(W(46), 128, 128, 4, 8, wr3);  pack(W(48), 128, 119, 4, 8, wr4);

  // ball query
  k_group<<<3072, 128, 0, stream>>>(spts, 3584, 2048, 1536, idxs);
  k_group<<<6144, 128, 0, stream>>>(lpts, 4096, 1024, 3072, idxl);

  // fused SA modules -> comb (B,4608,32) f16
  k_sa<<<6144, 128, 0, stream>>>(spts, 3584, 2048, 1536, fs, idxs,
                                 w1s, W(15), w2s, W(17), w3s, W(19), comb, 0);
  k_sa<<<12288, 128, 0, stream>>>(lpts, 4096, 1024, 3072, flm, idxl,
                                  w1l, W(21), w2l, W(23), w3l, W(25), comb, 1536);

  const int BIG = 1 << 30;  // identity row remap
  // match: comb -> 32 -> 2 ; out (B,4608,2) at +243712
  k_gemm<<<576, 128, 0, stream>>>(comb, 32, 1, wm1, 2, W(27), 32, BIG, 0, 0,
                                  0, nullptr, hbuf, nullptr, 0, 0, 0);
  k_gemm<<<576, 128, 0, stream>>>(hbuf, 32, 1, wm2, 1, W(29), 2, BIG, 0, 0,
                                  2, out + 243712, nullptr, nullptr, 0, 0, 0);
  // match_sem ; out at +317440
  k_gemm<<<576, 128, 0, stream>>>(comb, 32, 1, wms1, 2, W(31), 32, BIG, 0, 0,
                                  0, nullptr, hbuf, nullptr, 0, 0, 0);
  k_gemm<<<576, 128, 0, stream>>>(hbuf, 32, 1, wms2, 1, W(33), 2, BIG, 0, 0,
                                  2, out + 317440, nullptr, nullptr, 0, 0, 0);
  // sf: surface rows of comb -> 32 -> 32, scatter into xfin cols [0,192)
  k_gemm<<<192, 128, 0, stream>>>(comb, 32, 1, wsa1, 2, W(35), 32, 1536, 4608, 0,
                                  0, nullptr, hbuf, nullptr, 0, 0, 0);
  k_gemm<<<192, 128, 0, stream>>>(hbuf, 32, 1, wsa2, 2, W(37), 32, BIG, 0, 0,
                                  4, nullptr, xfin, nullptr, 0, 6, 1536);
  // lf: line rows of comb -> 32 -> 32, scatter into xfin cols [192,576)
  k_gemm<<<384, 128, 0, stream>>>(comb, 32, 1, wla1, 2, W(39), 32, 3072, 4608, 1536,
                                  0, nullptr, hbuf, nullptr, 0, 0, 0);
  k_gemm<<<384, 128, 0, stream>>>(hbuf, 32, 1, wla2, 2, W(41), 32, BIG, 0, 0,
                                  4, nullptr, xfin, nullptr, 192, 12, 3072);
  // final: 576 ->128 (+agg residual) ->128 ->128 ->119 (chmajor out)
  k_gemm<<<32, 128, 0, stream>>>(xfin, 576, 18, wr1, 8, W(43), 128, BIG, 0, 0,
                                 5, nullptr, xa, agg, 0, 0, 0);
  k_gemm<<<32, 128, 0, stream>>>(xa, 128, 4, wr2, 8, W(45), 128, BIG, 0, 0,
                                 0, nullptr, xb, nullptr, 0, 0, 0);
  k_gemm<<<32, 128, 0, stream>>>(xb, 128, 4, wr3, 8, W(47), 128, BIG, 0, 0,
                                 0, nullptr, xa, nullptr, 0, 0, 0);
  k_gemm<<<32, 128, 0, stream>>>(xa, 128, 4, wr4, 8, W(49), 119, BIG, 0, 0,
                                 3, out, nullptr, nullptr, 0, 0, 0);
}
